// Attention_61065845015171
// MI455X (gfx1250) — matrix operand order repaired and checked by reading, not compiled, not run on hardware
//
#include <hip/hip_runtime.h>
#include <hip/hip_bf16.h>
#include <math.h>
#include <stdint.h>

// ---------------------------------------------------------------------------
// ALiBi attention block for MI455X (gfx1250, wave32, WMMA, async-to-LDS).
// Pipeline: wconv(bf16,T via LDS transpose) -> layernorm(bf16)
//           -> 3x WMMA GEMM (Q,K,Vt) with double-buffered async LDS staging
//           -> flash-attention (WMMA QK^T + online softmax + WMMA PV)
//           -> WMMA GEMM with Wo -> f32 out.
// ---------------------------------------------------------------------------

typedef __bf16 bf16_t;
typedef __attribute__((ext_vector_type(16))) __bf16 bf16x16;
typedef __attribute__((ext_vector_type(8)))  float  f32x8;

#define DIM_   1024
#define SEQ_   2048
#define BATCH_ 2
#define HEADS_ 16
#define DHEAD_ 64
#define ROWS_  (BATCH_ * SEQ_)   // 4096 token rows

static __device__ __forceinline__ f32x8 f8_zero() {
  f32x8 z;
  #pragma unroll
  for (int i = 0; i < 8; ++i) z[i] = 0.0f;
  return z;
}

// --- CDNA5 async copy: 32B per lane, global -> LDS, no data VGPRs ----------
// INST_OFFSET is added to both the LDS and global addresses (ISA 08 §4.4),
// so one immediate advances both sides.
static __device__ __forceinline__ void async_cp32B(unsigned lds_addr,
                                                   const void* gaddr) {
  asm volatile(
      "global_load_async_to_lds_b128 %0, %1, off\n"
      "global_load_async_to_lds_b128 %0, %1, off offset:16"
      :
      : "v"(lds_addr), "v"(gaddr)
      : "memory");
}
static __device__ __forceinline__ void wait_async0() {
  asm volatile("s_wait_asynccnt 0x0" ::: "memory");
}
// Low 32 bits of a flat LDS pointer are the LDS byte address (ISA 10.2).
static __device__ __forceinline__ unsigned lds_addr_of(const void* p) {
  return (unsigned)(uintptr_t)p;
}

// A fragment (16x32 MxK, bf16), row-major source with leading dim `ld`.
// Lanes 0-15 row M=lane hold K=0..7 / K=16..23; lanes 16-31 hold K=8..15 /
// K=24..31 (ISA 7.12.2).
static __device__ __forceinline__ bf16x16 frag_a(const bf16_t* p, int ld) {
  const int lane = threadIdx.x & 31;
  const int m    = lane & 15;
  const int kh   = lane >> 4;
  const bf16_t* row = p + m * ld + kh * 8;
  bf16x16 f;
  #pragma unroll
  for (int t = 0; t < 8; ++t) f[t] = row[t];
  #pragma unroll
  for (int t = 0; t < 8; ++t) f[8 + t] = row[16 + t];
  return f;
}

// B fragment (32x16 KxN, bf16); source stored N-major: p[n*ld + k].
// lane = column N (mod 16); lanes 0-15 hold K=0..7 / K=16..23, lanes 16-31 K=8..15 / K=24..31.
static __device__ __forceinline__ bf16x16 frag_b(const bf16_t* p, int ld) {
  const int lane = threadIdx.x & 31;
  const int n    = lane & 15;
  const int kh   = lane >> 4;
  const bf16_t* row = p + n * ld + kh * 8;
  bf16x16 f;
  #pragma unroll
  for (int t = 0; t < 8; ++t) f[t] = row[t];
  #pragma unroll
  for (int t = 0; t < 8; ++t) f[8 + t] = row[16 + t];
  return f;
}

static __device__ __forceinline__ f32x8 wmma_bf16(bf16x16 a, bf16x16 b, f32x8 c) {
  return __builtin_amdgcn_wmma_f32_16x16x32_bf16(
      false, a, false, b, (short)0, c, false, false);
}

static __device__ __forceinline__ void cp16(bf16_t* dst, const bf16_t* src) {
  *reinterpret_cast<float4*>(dst)     = *reinterpret_cast<const float4*>(src);
  *reinterpret_cast<float4*>(dst + 8) = *reinterpret_cast<const float4*>(src + 8);
}

// ---------------------------------------------------------------------------
// Stage 1: weight f32 -> bf16, transposed to [n][k] through LDS so both the
// global reads (64B/lane) and writes (32B/lane) are coalesced.
// ---------------------------------------------------------------------------
__global__ void wconv_kernel(const float* __restrict__ W, bf16_t* __restrict__ Wt) {
  __shared__ __align__(16) bf16_t sT[64 * 72];
  const int t  = threadIdx.x;
  const int k0 = (blockIdx.x & 15) * 64;   // input row block
  const int n0 = (blockIdx.x >> 4) * 64;   // input col block
  const int r  = t >> 2;
  const int c0 = (t & 3) * 16;
  const float* g = W + (size_t)(k0 + r) * DIM_ + n0 + c0;
  #pragma unroll
  for (int i = 0; i < 16; ++i) sT[(c0 + i) * 72 + r] = (bf16_t)g[i];
  __syncthreads();
  bf16_t* o = Wt + (size_t)(n0 + r) * DIM_ + k0 + c0;
  #pragma unroll
  for (int i = 0; i < 16; ++i) o[i] = sT[r * 72 + c0 + i];
}

// ---------------------------------------------------------------------------
// Stage 2: LayerNorm over DIM, output bf16.
// ---------------------------------------------------------------------------
__global__ void ln_kernel(const float* __restrict__ x,
                          const float* __restrict__ gamma,
                          const float* __restrict__ beta,
                          bf16_t* __restrict__ xn) {
  const int row  = blockIdx.x;
  const int tid  = threadIdx.x;
  const int wave = tid >> 5;
  const int lane = tid & 31;
  const float* xr = x + (size_t)row * DIM_;

  float v[4];
  float s = 0.0f, s2 = 0.0f;
  #pragma unroll
  for (int i = 0; i < 4; ++i) {
    const float t = xr[tid + i * 256];
    v[i] = t; s += t; s2 += t * t;
  }
  #pragma unroll
  for (int o = 1; o < 32; o <<= 1) {
    s  += __shfl_xor(s,  o, 32);
    s2 += __shfl_xor(s2, o, 32);
  }
  __shared__ float ws[8], ws2[8];
  if (lane == 0) { ws[wave] = s; ws2[wave] = s2; }
  __syncthreads();
  float S = 0.0f, S2 = 0.0f;
  #pragma unroll
  for (int w = 0; w < 8; ++w) { S += ws[w]; S2 += ws2[w]; }
  const float mu  = S * (1.0f / DIM_);
  const float var = S2 * (1.0f / DIM_) - mu * mu;
  const float inv = rsqrtf(var + 1e-5f);
  #pragma unroll
  for (int i = 0; i < 4; ++i) {
    const int c = tid + i * 256;
    xn[(size_t)row * DIM_ + c] = (bf16_t)((v[i] - mu) * inv * gamma[c] + beta[c]);
  }
}

// ---------------------------------------------------------------------------
// Stage 3/5: tiled WMMA GEMM with double-buffered async-to-LDS staging.
// C[M][N] = A[M][K] * Bt[n][k].  Block tile 128x128, BK=32, 8 waves
// (each 64x32 = 4x2 WMMA/K-step).  OUT_MODE 0: bf16 row-major;
// 1: bf16 head-transposed [b][n][s] via LDS transpose (for V);
// 2: f32 row-major (final output).
// ---------------------------------------------------------------------------
template <int OUT_MODE>
__global__ void gemm_bf16_kernel(const bf16_t* __restrict__ A,
                                 const bf16_t* __restrict__ Bt,
                                 void* __restrict__ Out,
                                 int M, int N, int Kd) {
  __shared__ __align__(16) bf16_t sA[2][128 * 40];  // pad 32->40 (banks)
  __shared__ __align__(16) bf16_t sB[2][128 * 40];

  const int tid  = threadIdx.x;
  const int wave = tid >> 5;
  const int lane = tid & 31;
  const int wm   = wave >> 2;   // 0..1 (64 rows)
  const int wn   = wave & 3;    // 0..3 (32 cols)
  const int m0   = blockIdx.y * 128;
  const int n0   = blockIdx.x * 128;

  f32x8 acc[4][2];
  #pragma unroll
  for (int i = 0; i < 4; ++i)
    #pragma unroll
    for (int j = 0; j < 2; ++j) acc[i][j] = f8_zero();

  // per-thread 32B staging chunk
  const int r  = tid >> 1;          // 0..127
  const int c0 = (tid & 1) * 16;    // 0 or 16
  const bf16_t* gA = A  + (size_t)(m0 + r) * Kd + c0;
  const bf16_t* gB = Bt + (size_t)(n0 + r) * Kd + c0;
  const unsigned uA = lds_addr_of(&sA[0][0]) + (unsigned)(r * 40 + c0) * 2u;
  const unsigned uB = lds_addr_of(&sB[0][0]) + (unsigned)(r * 40 + c0) * 2u;
  const unsigned BUFB = 128u * 40u * 2u;   // 10240 B per buffer

  // preload tile 0
  async_cp32B(uA, gA);
  async_cp32B(uB, gB);
  wait_async0();
  __syncthreads();

  for (int k0 = 0; k0 < Kd; k0 += 32) {
    const int cur = (k0 >> 5) & 1;
    if (k0 + 32 < Kd) {              // kick off next tile while we compute
      const unsigned nb = (unsigned)(cur ^ 1) * BUFB;
      async_cp32B(uA + nb, gA + k0 + 32);
      async_cp32B(uB + nb, gB + k0 + 32);
    }
    const bf16_t* a = &sA[cur][0];
    const bf16_t* bT = &sB[cur][0];

    bf16x16 af[4], bfg[2];
    #pragma unroll
    for (int ms = 0; ms < 4; ++ms)
      af[ms] = frag_a(a + (wm * 64 + ms * 16) * 40, 40);
    #pragma unroll
    for (int ns = 0; ns < 2; ++ns)
      bfg[ns] = frag_b(bT + (wn * 32 + ns * 16) * 40, 40);
    #pragma unroll
    for (int ms = 0; ms < 4; ++ms)
      #pragma unroll
      for (int ns = 0; ns < 2; ++ns)
        acc[ms][ns] = wmma_bf16(af[ms], bfg[ns], acc[ms][ns]);

    wait_async0();     // next tile landed in LDS
    __syncthreads();   // everyone done reading cur
  }

  const int lane_n = lane & 15;
  const int mhi    = (lane >> 4) * 8;

  if (OUT_MODE == 1) {
    // head-transposed store: C-layout tile -> LDS (n-major) -> 32B row stores
    bf16_t* st = &sA[0][0] + wave * (16 * 24);   // per-wave scratch, 48B rows
    bf16_t* outp = (bf16_t*)Out;
    #pragma unroll
    for (int ms = 0; ms < 4; ++ms) {
      #pragma unroll
      for (int ns = 0; ns < 2; ++ns) {
        #pragma unroll
        for (int e = 0; e < 8; ++e)
          st[lane_n * 24 + mhi + e] = (bf16_t)acc[ms][ns][e];
        if (lane < 16) {   // one lane per output row n, contiguous along s
          const int gn  = n0 + wn * 32 + ns * 16 + lane;
          const int gmb = m0 + wm * 64 + ms * 16;
          const int bb  = gmb >> 11;
          const int s0v = gmb & (SEQ_ - 1);
          bf16_t* o = outp + (size_t)bb * N * SEQ_ + (size_t)gn * SEQ_ + s0v;
          cp16(o, st + lane * 24);
        }
      }
    }
  } else {
    #pragma unroll
    for (int ms = 0; ms < 4; ++ms) {
      #pragma unroll
      for (int ns = 0; ns < 2; ++ns) {
        const int gm_base = m0 + wm * 64 + ms * 16 + mhi;
        const int gn      = n0 + wn * 32 + ns * 16 + lane_n;
        #pragma unroll
        for (int e = 0; e < 8; ++e) {
          const int gm  = gm_base + e;
          const float v = acc[ms][ns][e];
          if (OUT_MODE == 0) ((bf16_t*)Out)[(size_t)gm * N + gn] = (bf16_t)v;
          else               ((float*)Out)[(size_t)gm * N + gn]  = v;
        }
      }
    }
  }
}

// ---------------------------------------------------------------------------
// Stage 4: flash attention with ALiBi, double-buffered async K/V staging.
// Block = 64 query rows of one (b,h); 4 waves x 16 rows.  Per 32-key step:
// 4 WMMA (QK^T, K=64) + online softmax + 4 WMMA (PV).
// ---------------------------------------------------------------------------
__global__ void attn_kernel(const bf16_t* __restrict__ Q,
                            const bf16_t* __restrict__ K,
                            const bf16_t* __restrict__ Vt,   // [b][c][s]
                            bf16_t* __restrict__ O) {
  const int qt = blockIdx.x;
  const int h  = blockIdx.y;
  const int b  = blockIdx.z;
  const int tid  = threadIdx.x;
  const int wave = tid >> 5;
  const int lane = tid & 31;
  const int qbase = qt * 64 + wave * 16;

  __shared__ __align__(16) bf16_t sK[2][32 * 72];     // [j][d], pad 64->72
  __shared__ __align__(16) bf16_t sV[2][64 * 40];     // [d][j], pad 32->40
  __shared__ __align__(16) bf16_t sP[4][16 * 40];     // per-wave P transpose

  // Q strip (16x64) stays in registers for the whole pass.
  const bf16_t* Qrow = Q + ((size_t)(b * SEQ_ + qbase)) * DIM_ + h * DHEAD_;
  bf16x16 qf[2];
  qf[0] = frag_a(Qrow,      DIM_);
  qf[1] = frag_a(Qrow + 32, DIM_);

  const float slope = exp2f(-8.0f / (float)(HEADS_ - h));
  const int lane_n = lane & 15;
  const int hi8    = (lane >> 4) * 8;

  float m_run[8], l_run[8];
  f32x8 oacc[4];
  #pragma unroll
  for (int e = 0; e < 8; ++e) { m_run[e] = -1e30f; l_run[e] = 0.0f; }
  #pragma unroll
  for (int dn = 0; dn < 4; ++dn) oacc[dn] = f8_zero();

  int gq[8];
  #pragma unroll
  for (int e = 0; e < 8; ++e) gq[e] = qbase + hi8 + e;

  // per-thread async staging chunks
  const int rk = tid >> 2, ck = (tid & 3) * 16;   // K tile 32x64
  const int dv = tid >> 1, cv = (tid & 1) * 16;   // V tile 64x32
  const bf16_t* gK = K  + ((size_t)(b * SEQ_ + rk)) * DIM_ + h * DHEAD_ + ck;
  const bf16_t* gV = Vt + ((size_t)b * DIM_ + h * DHEAD_ + dv) * SEQ_ + cv;
  const unsigned uK = lds_addr_of(&sK[0][0]) + (unsigned)(rk * 72 + ck) * 2u;
  const unsigned uV = lds_addr_of(&sV[0][0]) + (unsigned)(dv * 40 + cv) * 2u;
  const unsigned KB = 32u * 72u * 2u;
  const unsigned VB = 64u * 40u * 2u;

  async_cp32B(uK, gK);
  async_cp32B(uV, gV);
  wait_async0();
  __syncthreads();

  for (int j0 = 0; j0 < SEQ_; j0 += 32) {
    const int cur = (j0 >> 5) & 1;
    if (j0 + 32 < SEQ_) {
      const unsigned nx = (unsigned)(cur ^ 1);
      async_cp32B(uK + nx * KB, gK + (size_t)(j0 + 32) * DIM_);
      async_cp32B(uV + nx * VB, gV + (j0 + 32));
    }
    const bf16_t* kb = &sK[cur][0];
    const bf16_t* vb = &sV[cur][0];

    // scores: S(16x32) = Q(16x64) . K^T, K split 32+32
    f32x8 s0 = f8_zero(), s1 = f8_zero();
    s0 = wmma_bf16(qf[0], frag_b(kb + 0 * 72 + 0,  72), s0);
    s0 = wmma_bf16(qf[1], frag_b(kb + 0 * 72 + 32, 72), s0);
    s1 = wmma_bf16(qf[0], frag_b(kb + 16 * 72 + 0,  72), s1);
    s1 = wmma_bf16(qf[1], frag_b(kb + 16 * 72 + 32, 72), s1);

    // ALiBi + online softmax (rows live across 16-lane halves in C-layout)
    float alpha[8];
    #pragma unroll
    for (int e = 0; e < 8; ++e) {
      const int gi = gq[e];
      const int ja = j0 + lane_n, jb = j0 + 16 + lane_n;
      float sa = s0[e] * 0.125f - slope * fabsf((float)(gi - ja));
      float sb = s1[e] * 0.125f - slope * fabsf((float)(gi - jb));
      float mx = fmaxf(sa, sb);
      #pragma unroll
      for (int o = 1; o < 16; o <<= 1) mx = fmaxf(mx, __shfl_xor(mx, o, 32));
      const float mnew = fmaxf(m_run[e], mx);
      const float a_   = __expf(m_run[e] - mnew);
      const float pa   = __expf(sa - mnew);
      const float pb   = __expf(sb - mnew);
      float rs = pa + pb;
      #pragma unroll
      for (int o = 1; o < 16; o <<= 1) rs += __shfl_xor(rs, o, 32);
      l_run[e] = l_run[e] * a_ + rs;
      m_run[e] = mnew;
      alpha[e] = a_;
      sP[wave][(hi8 + e) * 40 + lane_n]      = (bf16_t)pa;
      sP[wave][(hi8 + e) * 40 + 16 + lane_n] = (bf16_t)pb;
    }
    #pragma unroll
    for (int dn = 0; dn < 4; ++dn)
      #pragma unroll
      for (int e = 0; e < 8; ++e) oacc[dn][e] *= alpha[e];

    const bf16x16 pf = frag_a(&sP[wave][0], 40);
    #pragma unroll
    for (int dn = 0; dn < 4; ++dn)
      oacc[dn] = wmma_bf16(pf, frag_b(vb + (dn * 16) * 40, 40), oacc[dn]);

    wait_async0();
    __syncthreads();
  }

  #pragma unroll
  for (int e = 0; e < 8; ++e) {
    const float inv = 1.0f / l_run[e];
    const size_t rowoff = ((size_t)(b * SEQ_ + gq[e])) * DIM_ + h * DHEAD_;
    #pragma unroll
    for (int dn = 0; dn < 4; ++dn)
      O[rowoff + dn * 16 + lane_n] = (bf16_t)(oacc[dn][e] * inv);
  }
}

// ---------------------------------------------------------------------------
// Host driver.
// ---------------------------------------------------------------------------
extern "C" void kernel_launch(void* const* d_in, const int* in_sizes, int n_in,
                              void* d_out, int out_size, void* d_ws, size_t ws_size,
                              hipStream_t stream) {
  (void)in_sizes; (void)n_in; (void)out_size; (void)ws_size;
  const float* x     = (const float*)d_in[0];
  const float* gamma = (const float*)d_in[1];
  const float* beta  = (const float*)d_in[2];
  const float* Wq    = (const float*)d_in[3];
  const float* Wk    = (const float*)d_in[4];
  const float* Wv    = (const float*)d_in[5];
  const float* Wo    = (const float*)d_in[6];

  char* ws = (char*)d_ws;
  const size_t MiB = (size_t)1 << 20;
  bf16_t* xn  = (bf16_t*)(ws + 0 * MiB);   // 8 MiB  [4096][1024]
  bf16_t* WqT = (bf16_t*)(ws + 8 * MiB);   // 2 MiB  [n][k]
  bf16_t* WkT = (bf16_t*)(ws + 10 * MiB);
  bf16_t* WvT = (bf16_t*)(ws + 12 * MiB);
  bf16_t* WoT = (bf16_t*)(ws + 14 * MiB);
  bf16_t* Qb  = (bf16_t*)(ws + 16 * MiB);  // 8 MiB  [4096][1024]
  bf16_t* Kb  = (bf16_t*)(ws + 24 * MiB);  // 8 MiB
  bf16_t* Vtb = (bf16_t*)(ws + 32 * MiB);  // 8 MiB  [b][c][s]
  bf16_t* AO  = (bf16_t*)(ws + 40 * MiB);  // 8 MiB

  const dim3 gW(16 * 16);                  // 64x64 transpose tiles
  wconv_kernel<<<gW, 256, 0, stream>>>(Wq, WqT);
  wconv_kernel<<<gW, 256, 0, stream>>>(Wk, WkT);
  wconv_kernel<<<gW, 256, 0, stream>>>(Wv, WvT);
  wconv_kernel<<<gW, 256, 0, stream>>>(Wo, WoT);

  ln_kernel<<<ROWS_, 256, 0, stream>>>(x, gamma, beta, xn);

  const dim3 gGemm(DIM_ / 128, ROWS_ / 128);  // (8, 32)
  gemm_bf16_kernel<0><<<gGemm, 256, 0, stream>>>(xn, WqT, (void*)Qb,  ROWS_, DIM_, DIM_);
  gemm_bf16_kernel<0><<<gGemm, 256, 0, stream>>>(xn, WkT, (void*)Kb,  ROWS_, DIM_, DIM_);
  gemm_bf16_kernel<1><<<gGemm, 256, 0, stream>>>(xn, WvT, (void*)Vtb, ROWS_, DIM_, DIM_);

  attn_kernel<<<dim3(SEQ_ / 64, HEADS_, BATCH_), 128, 0, stream>>>(Qb, Kb, Vtb, AO);

  gemm_bf16_kernel<2><<<gGemm, 256, 0, stream>>>(AO, WoT, d_out, ROWS_, DIM_, DIM_);
}
